// MyModel_80573586473257
// MI455X (gfx1250) — matrix operand order repaired and checked by reading, not compiled, not run on hardware
//
#include <hip/hip_runtime.h>
#include <hip/hip_bf16.h>

// ---------------------------------------------------------------------------
// DGCNN forward for MI455X (gfx1250, wave32).
// GEMMs -> v_wmma_f32_16x16x32_f16, templated per A-gather mode, with
// double-buffered LDS staging (1 barrier / K-step). Fragments are loaded
// into registers BEFORE next-slab staging is issued, so the DScnt drain
// covers only fragment loads and the 4 WMMAs run back-to-back while the
// next slab's global loads are in flight.
// BN(batch stats) folded to per-channel affine applied by the consumer.
// ---------------------------------------------------------------------------

typedef __attribute__((ext_vector_type(16))) _Float16 v16h;
typedef __attribute__((ext_vector_type(8)))  float    v8f;

#define NPTS  512
#define EPSF  1e-5f
#define LDA   40   // A row stride in halves (80B: 16B-aligned, skewed)
#define LDBF  24   // B fragment lane stride in halves (48B: 16B-aligned, skewed)

#define MODE_DENSE  0
#define MODE_EDGE   1
#define MODE_CONCAT 2

// ------------------------------- utility kernels ---------------------------

__global__ void cvt_f32_f16_kernel(const float* __restrict__ in,
                                   _Float16* __restrict__ out, int n) {
  int t = blockIdx.x * blockDim.x + threadIdx.x;
  if (t < n) out[t] = (_Float16)in[t];
}

__global__ void zero_f32_kernel(float* __restrict__ p, int n) {
  int t = blockIdx.x * blockDim.x + threadIdx.x;
  if (t < n) p[t] = 0.f;
}

// BN finalize: scale = g*rsqrt(var+eps), shift = be - mu*scale  (biased var)
__global__ void bn_finalize_kernel(const float* __restrict__ sum,
                                   const float* __restrict__ sumsq,
                                   const float* __restrict__ g,
                                   const float* __restrict__ be,
                                   float* __restrict__ scl,
                                   float* __restrict__ shf,
                                   int C, float invM) {
  int c = blockIdx.x * blockDim.x + threadIdx.x;
  if (c >= C) return;
  float mu  = sum[c] * invM;
  float var = sumsq[c] * invM - mu * mu;
  float a   = g[c] * rsqrtf(var + EPSF);
  scl[c] = a;
  shf[c] = be[c] - mu * a;
}

// ------------------------------- kNN ---------------------------------------
template <int D, int K, typename T>
__global__ void knn_kernel(const T* __restrict__ X, int* __restrict__ idx,
                           int BN) {
  int t = blockIdx.x * blockDim.x + threadIdx.x;
  if (t >= BN) return;
  int b = t >> 9, n = t & (NPTS - 1);
  const T* xb = X + (size_t)b * NPTS * D;
  float bestd[K]; int besti[K];
#pragma unroll
  for (int i = 0; i < K; i++) { bestd[i] = 1e30f; besti[i] = 0; }
  for (int m = 0; m < NPTS; m++) {
    if (m == n) continue;                       // loop=False: self excluded
    float d2 = 0.f;
    for (int d = 0; d < D; d++) {
      float df = (float)xb[n * D + d] - (float)xb[m * D + d];
      d2 = fmaf(df, df, d2);
    }
    if (d2 < bestd[K - 1]) {
      bestd[K - 1] = d2; besti[K - 1] = m;
#pragma unroll
      for (int i = K - 1; i > 0; i--) {
        if (bestd[i] < bestd[i - 1]) {
          float td = bestd[i]; bestd[i] = bestd[i - 1]; bestd[i - 1] = td;
          int ti = besti[i]; besti[i] = besti[i - 1]; besti[i - 1] = ti;
        }
      }
    }
  }
#pragma unroll
  for (int i = 0; i < K; i++) idx[t * K + i] = besti[i];
}

// ------------------------------- WMMA GEMM ---------------------------------
// Y[M,Cout] = ReLU(A[M,Cin] @ W^T + bias), Y stored f16, per-channel
// sum/sumsq accumulated for BatchNorm. A gathered per MODE (compile-time).
// Block = 128 threads = 4 waves; 64x64 tile; wave w owns rows [16w,16w+16).

template <int MODE, bool NORM>
__global__ void __launch_bounds__(128)
gemm_bn_wmma(int M, int Cin, int Cout,
             const _Float16* __restrict__ A0, const _Float16* __restrict__ A1,
             const _Float16* __restrict__ A2, int Ca, int Cb,
             const int* __restrict__ idx, int klog,
             const float* __restrict__ nscl, const float* __restrict__ nshf,
             const _Float16* __restrict__ W, const float* __restrict__ bias,
             _Float16* __restrict__ Y,
             float* __restrict__ csum, float* __restrict__ csq) {
  __shared__ __align__(16) _Float16 Alds[2][64 * LDA];
  __shared__ __align__(16) _Float16 Bf[2][4 * 32 * LDBF];   // fragment-major

  int m0 = blockIdx.x * 64;
  int n0 = blockIdx.y * 64;
  int tid = threadIdx.x, lane = tid & 31, wv = tid >> 5;

  // ---- per-thread A staging assignment: row aml, 16-col chunk akc ----
  int aml = tid >> 1;
  int akc = (tid & 1) << 4;
  int agm = m0 + aml;
  bool arow_ok = agm < M;
  const _Float16* rowI = nullptr;   // dense/edge center row
  const _Float16* rowJ = nullptr;   // edge neighbor row
  if (arow_ok) {
    if (MODE == MODE_EDGE) {
      int bn = agm >> klog;                   // b*NPTS + n
      int b  = bn >> 9;                       // NPTS = 512
      int j  = idx[agm];
      rowI = A0 + bn * Ca;
      rowJ = A0 + (b * NPTS + j) * Ca;
    } else if (MODE == MODE_DENSE) {
      rowI = A0 + agm * Cin;
    }
  }

  // ---- staging routines (16 contiguous halves / thread for A; 2x8 for B) --
  auto stageA = [&](int k0, _Float16* Ab) {
    _Float16* dst = &Ab[aml * LDA + akc];
    int cbase = k0 + akc;
    if (!arow_ok) {
#pragma unroll
      for (int e = 0; e < 16; e++) dst[e] = (_Float16)0.f;
      return;
    }
    if (MODE == MODE_DENSE) {
      if (cbase + 16 <= Cin) {
        if (NORM) {
#pragma unroll
          for (int e = 0; e < 16; e++)
            dst[e] = (_Float16)((float)rowI[cbase + e] * nscl[cbase + e] +
                                nshf[cbase + e]);
        } else {
#pragma unroll
          for (int e = 0; e < 16; e++) dst[e] = rowI[cbase + e];
        }
      } else {
#pragma unroll
        for (int e = 0; e < 16; e++) {
          int c = cbase + e;
          float v = 0.f;
          if (c < Cin) {
            v = (float)rowI[c];
            if (NORM) v = v * nscl[c] + nshf[c];
          }
          dst[e] = (_Float16)v;
        }
      }
    } else if (MODE == MODE_EDGE) {
      int C = Ca;
      if (cbase + 16 <= C) {                           // fully in x_i
#pragma unroll
        for (int e = 0; e < 16; e++) dst[e] = rowI[cbase + e];
      } else if (cbase >= C && cbase + 16 <= 2 * C) {  // fully in x_j - x_i
        int cc = cbase - C;
#pragma unroll
        for (int e = 0; e < 16; e++)
          dst[e] = (_Float16)((float)rowJ[cc + e] - (float)rowI[cc + e]);
      } else {                                         // straddle / pad (conv1)
#pragma unroll
        for (int e = 0; e < 16; e++) {
          int c = cbase + e;
          float v = 0.f;
          if (c < C) v = (float)rowI[c];
          else if (c < 2 * C) v = (float)rowJ[c - C] - (float)rowI[c - C];
          dst[e] = (_Float16)v;
        }
      }
    } else {  // MODE_CONCAT: Ca,Cb multiples of 16 -> chunks never straddle
      const _Float16* src = nullptr;
      if (cbase + 16 <= Ca) src = A0 + agm * Ca + cbase;
      else if (cbase + 16 <= Ca + Cb) src = A1 + agm * Cb + (cbase - Ca);
      else if (cbase + 16 <= Cin)
        src = A2 + agm * (Cin - Ca - Cb) + (cbase - Ca - Cb);
      if (src) {
#pragma unroll
        for (int e = 0; e < 16; e++) dst[e] = src[e];
      } else {
#pragma unroll
        for (int e = 0; e < 16; e++) dst[e] = (_Float16)0.f;
      }
    }
  };
  // element (k,n) -> subtile s=n/16, lane = n%16 + 16*(k/16), elem = k%16
  auto stageB = [&](int k0, _Float16* Bb) {
#pragma unroll
    for (int q0 = 0; q0 < 2; q0++) {
      int q   = tid + (q0 << 7);     // 0..255
      int n   = q >> 2;              // 0..63
      int kc8 = (q & 3) << 3;        // 0,8,16,24
      int gn  = n0 + n;
      int ln  = (n & 15) + ((kc8 >> 4) << 4);
      _Float16* dst = &Bb[((n >> 4) * 32 + ln) * LDBF + (kc8 & 15)];
      int gkb = k0 + kc8;
      if (gn < Cout && gkb + 8 <= Cin) {
        const _Float16* src = W + gn * Cin + gkb;
#pragma unroll
        for (int e = 0; e < 8; e++) dst[e] = src[e];
      } else {
#pragma unroll
        for (int e = 0; e < 8; e++) {
          int gk = gkb + e;
          dst[e] = (gn < Cout && gk < Cin) ? W[gn * Cin + gk] : (_Float16)0.f;
        }
      }
    }
  };

  v8f acc[4];
#pragma unroll
  for (int s = 0; s < 4; s++)
#pragma unroll
    for (int r = 0; r < 8; r++) acc[s][r] = 0.f;

  // ---- double-buffered main loop: one barrier per K-step ----
  int nk = (Cin + 31) >> 5;
  stageA(0, Alds[0]);
  stageB(0, Bf[0]);
  for (int it = 0; it < nk; ++it) {
    __syncthreads();                 // staged[cur] ready; prev reads done
    int cur = it & 1;

    // (1) fragment loads from LDS into registers (the only DS deps of WMMA)
    const _Float16* ap = &Alds[cur][(wv * 16 + (lane & 15)) * LDA];
    int akb = (lane >> 4) << 4;
    v16h af;
#pragma unroll
    for (int e = 0; e < 8; e++) af[e] = ap[akb + e];
#pragma unroll
    for (int e = 0; e < 8; e++) af[8 + e] = ap[akb + 8 + e];
    v16h bfr[4];
#pragma unroll
    for (int s = 0; s < 4; s++) {
      const _Float16* bp = &Bf[cur][(s * 32 + lane) * LDBF];
#pragma unroll
      for (int e = 0; e < 16; e++) bfr[s][e] = bp[e];
    }

    // (2) issue next K-slab staging (global loads + LDS stores) now; its
    //     latency is hidden behind the WMMA chain below.
    if (it + 1 < nk) {
      stageA((it + 1) << 5, Alds[cur ^ 1]);
      stageB((it + 1) << 5, Bf[cur ^ 1]);
    }

    // (3) matrix ops back-to-back
#pragma unroll
    for (int s = 0; s < 4; s++)
      acc[s] = __builtin_amdgcn_wmma_f32_16x16x32_f16(
          false, af, false, bfr[s], (short)0, acc[s], false, false);
  }

  // epilogue: bias + ReLU, store f16, per-channel sum/sumsq for BN
  int nl    = lane & 15;
  int mrow0 = wv * 16 + ((lane >> 4) << 3);   // D layout: M = r (+8 upper half)
#pragma unroll
  for (int s = 0; s < 4; s++) {
    int gn = n0 + s * 16 + nl;
    if (gn >= Cout) continue;
    float bsv = bias ? bias[gn] : 0.f;
    float lsum = 0.f, lsq = 0.f;
#pragma unroll
    for (int r = 0; r < 8; r++) {
      int gm = m0 + mrow0 + r;
      if (gm < M) {
        float v = acc[s][r] + bsv;
        v = v > 0.f ? v : 0.f;                 // ReLU (all WMMA layers)
        Y[gm * Cout + gn] = (_Float16)v;
        lsum += v; lsq += v * v;
      }
    }
    if (csum) {
      atomicAdd(&csum[gn], lsum);
      atomicAdd(&csq[gn], lsq);
    }
  }
}

// ------------------------- reductions with BN folded ------------------------
// C is a power of two; clog = log2(C).

__global__ void maxk_kernel(const _Float16* __restrict__ Y,
                            const float* __restrict__ scl,
                            const float* __restrict__ shf,
                            _Float16* __restrict__ X,
                            int total, int klog, int clog) {
  int t = blockIdx.x * blockDim.x + threadIdx.x;
  if (t >= total) return;
  int bn = t >> clog, c = t & ((1 << clog) - 1);
  int C = 1 << clog, KNBR = 1 << klog;
  float a = scl[c], bb = shf[c];
  float best = -1e30f;
  const _Float16* src = Y + ((bn << klog) << clog) + c;
  for (int kk = 0; kk < KNBR; kk++)
    best = fmaxf(best, (float)src[kk * C] * a + bb);
  X[t] = (_Float16)best;
}

__global__ void maxpool_n_kernel(const _Float16* __restrict__ H,
                                 const float* __restrict__ scl,
                                 const float* __restrict__ shf,
                                 _Float16* __restrict__ G,
                                 int total, int clog) {
  int t = blockIdx.x * blockDim.x + threadIdx.x;
  if (t >= total) return;
  int b = t >> clog, c = t & ((1 << clog) - 1);
  int C = 1 << clog;
  float a = scl[c], bb = shf[c];
  float best = -1e30f;
  const _Float16* src = H + ((b * NPTS) << clog) + c;
  for (int n = 0; n < NPTS; n++)
    best = fmaxf(best, (float)src[n * C] * a + bb);
  G[t] = (_Float16)best;
}

__global__ void final_fc_kernel(const _Float16* __restrict__ Yf2,
                                const float* __restrict__ scl,
                                const float* __restrict__ shf,
                                const float* __restrict__ W,
                                const float* __restrict__ b,
                                float* __restrict__ out, int B, int C) {
  int bi = blockIdx.x * blockDim.x + threadIdx.x;
  if (bi >= B) return;
  float dot = 0.f;
  for (int c = 0; c < C; c++)
    dot += ((float)Yf2[bi * C + c] * scl[c] + shf[c]) * W[c];
  dot += b[0];
  out[bi] = 1.f / (1.f + expf(-dot));
}

// ------------------------------- host side ----------------------------------

extern "C" void kernel_launch(void* const* d_in, const int* in_sizes, int n_in,
                              void* d_out, int out_size, void* d_ws, size_t ws_size,
                              hipStream_t stream) {
  (void)in_sizes; (void)n_in; (void)out_size; (void)ws_size;
  const int B = 64, N = 512;
  const int BN = B * N;

  const float* x    = (const float*)d_in[0];
  const float* pos  = (const float*)d_in[1];
  const float* c1W1 = (const float*)d_in[2];  const float* c1b1 = (const float*)d_in[3];
  const float* c1g1 = (const float*)d_in[4];  const float* c1e1 = (const float*)d_in[5];
  const float* c1W2 = (const float*)d_in[6];  const float* c1b2 = (const float*)d_in[7];
  const float* c1g2 = (const float*)d_in[8];  const float* c1e2 = (const float*)d_in[9];
  const float* c2W  = (const float*)d_in[10]; const float* c2b  = (const float*)d_in[11];
  const float* c2g  = (const float*)d_in[12]; const float* c2e  = (const float*)d_in[13];
  const float* c3W  = (const float*)d_in[14]; const float* c3b  = (const float*)d_in[15];
  const float* c3g  = (const float*)d_in[16]; const float* c3e  = (const float*)d_in[17];
  const float* l1W  = (const float*)d_in[18]; const float* l1b  = (const float*)d_in[19];
  const float* l1g  = (const float*)d_in[20]; const float* l1e  = (const float*)d_in[21];
  const float* f1W  = (const float*)d_in[22]; const float* f1b  = (const float*)d_in[23];
  const float* f1g  = (const float*)d_in[24]; const float* f1e  = (const float*)d_in[25];
  const float* f2W  = (const float*)d_in[26]; const float* f2b  = (const float*)d_in[27];
  const float* f2g  = (const float*)d_in[28]; const float* f2e  = (const float*)d_in[29];
  const float* f3W  = (const float*)d_in[30]; const float* f3b  = (const float*)d_in[31];

  // --- workspace carve-up (deterministic) ---
  char* wsb = (char*)d_ws;
  size_t off = 0;
  auto ALLOC = [&](size_t bytes) -> char* {
    char* p = wsb + off; off += (bytes + 255) & ~(size_t)255; return p;
  };
  _Float16* xh    = (_Float16*)ALLOC((size_t)BN * 3 * 2);
  _Float16* c1W1h = (_Float16*)ALLOC(64 * 6 * 2);
  _Float16* c1W2h = (_Float16*)ALLOC(128 * 64 * 2);
  _Float16* c2Wh  = (_Float16*)ALLOC(256 * 256 * 2);
  _Float16* c3Wh  = (_Float16*)ALLOC((size_t)512 * 512 * 2);
  _Float16* l1Wh  = (_Float16*)ALLOC((size_t)1024 * 896 * 2);
  _Float16* f1Wh  = (_Float16*)ALLOC((size_t)512 * 1024 * 2);
  _Float16* f2Wh  = (_Float16*)ALLOC(256 * 512 * 2);
  int* idx1 = (int*)ALLOC((size_t)BN * 8 * 4);
  int* idx2 = (int*)ALLOC((size_t)BN * 4 * 4);
  int* idx3 = (int*)ALLOC((size_t)BN * 4 * 4);
  char* stats0 = wsb + off;                    // [sum|sumsq|scl|shf] per layer
  float* sA = (float*)ALLOC(4 * 64 * 4);
  float* sB = (float*)ALLOC(4 * 128 * 4);
  float* sC = (float*)ALLOC(4 * 256 * 4);
  float* sD = (float*)ALLOC(4 * 512 * 4);
  float* sE = (float*)ALLOC(4 * 1024 * 4);
  float* sF = (float*)ALLOC(4 * 512 * 4);
  float* sG = (float*)ALLOC(4 * 256 * 4);
  size_t statsBytes = (size_t)((wsb + off) - stats0);
  _Float16* x1h  = (_Float16*)ALLOC((size_t)BN * 128 * 2);
  _Float16* x2h  = (_Float16*)ALLOC((size_t)BN * 256 * 2);
  _Float16* x3h  = (_Float16*)ALLOC((size_t)BN * 512 * 2);
  _Float16* gbh  = (_Float16*)ALLOC((size_t)B * 1024 * 2);
  _Float16* yf1  = (_Float16*)ALLOC((size_t)B * 512 * 2);
  _Float16* yf2  = (_Float16*)ALLOC((size_t)B * 256 * 2);
  char* regA = ALLOC((size_t)134217728);       // y1b (67MB) then y3 (134MB)
  char* regB = ALLOC((size_t)67108864);        // y1a (33MB), y2 (67MB), h (67MB)
  _Float16* y1a = (_Float16*)regB;
  _Float16* y1b = (_Float16*)regA;
  _Float16* y2  = (_Float16*)regB;
  _Float16* y3  = (_Float16*)regA;
  _Float16* hb  = (_Float16*)regB;

  auto CVT = [&](const float* s, _Float16* d, int n) {
    cvt_f32_f16_kernel<<<(n + 255) / 256, 256, 0, stream>>>(s, d, n);
  };
  auto BNFIN = [&](float* st, const float* g, const float* be, int C, float invM) {
    bn_finalize_kernel<<<(C + 255) / 256, 256, 0, stream>>>(
        st, st + C, g, be, st + 2 * C, st + 3 * C, C, invM);
  };
  auto GRID = [](int M, int Cout) {
    return dim3((unsigned)((M + 63) / 64), (unsigned)((Cout + 63) / 64));
  };

  // 0) conversions + zero stats
  CVT(x, xh, BN * 3);
  CVT(c1W1, c1W1h, 64 * 6);         CVT(c1W2, c1W2h, 128 * 64);
  CVT(c2W, c2Wh, 256 * 256);        CVT(c3W, c3Wh, 512 * 512);
  CVT(l1W, l1Wh, 1024 * 896);       CVT(f1W, f1Wh, 512 * 1024);
  CVT(f2W, f2Wh, 256 * 512);
  {
    int n = (int)(statsBytes / 4);
    zero_f32_kernel<<<(n + 255) / 256, 256, 0, stream>>>((float*)stats0, n);
  }

  // 1) conv1: kNN(pos,8) -> EdgeConv MLP [6->64->128] -> max_k
  knn_kernel<3, 8, float><<<(BN + 127) / 128, 128, 0, stream>>>(pos, idx1, BN);
  gemm_bn_wmma<MODE_EDGE, false><<<GRID(BN * 8, 64), 128, 0, stream>>>(
      BN * 8, 6, 64, xh, nullptr, nullptr, 3, 0, idx1, 3, nullptr, nullptr,
      c1W1h, c1b1, y1a, sA, sA + 64);
  BNFIN(sA, c1g1, c1e1, 64, 1.f / (float)(BN * 8));
  gemm_bn_wmma<MODE_DENSE, true><<<GRID(BN * 8, 128), 128, 0, stream>>>(
      BN * 8, 64, 128, y1a, nullptr, nullptr, 0, 0, nullptr, 0,
      sA + 2 * 64, sA + 3 * 64, c1W2h, c1b2, y1b, sB, sB + 128);
  BNFIN(sB, c1g2, c1e2, 128, 1.f / (float)(BN * 8));
  maxk_kernel<<<(BN * 128 + 255) / 256, 256, 0, stream>>>(
      y1b, sB + 2 * 128, sB + 3 * 128, x1h, BN * 128, 3, 7);

  // 2) conv2: kNN(x1,4) -> MLP [256->256] -> max_k
  knn_kernel<128, 4, _Float16><<<(BN + 127) / 128, 128, 0, stream>>>(x1h, idx2, BN);
  gemm_bn_wmma<MODE_EDGE, false><<<GRID(BN * 4, 256), 128, 0, stream>>>(
      BN * 4, 256, 256, x1h, nullptr, nullptr, 128, 0, idx2, 2, nullptr, nullptr,
      c2Wh, c2b, y2, sC, sC + 256);
  BNFIN(sC, c2g, c2e, 256, 1.f / (float)(BN * 4));
  maxk_kernel<<<(BN * 256 + 255) / 256, 256, 0, stream>>>(
      y2, sC + 2 * 256, sC + 3 * 256, x2h, BN * 256, 2, 8);

  // 3) conv3: kNN(x2,4) -> MLP [512->512] -> max_k
  knn_kernel<256, 4, _Float16><<<(BN + 127) / 128, 128, 0, stream>>>(x2h, idx3, BN);
  gemm_bn_wmma<MODE_EDGE, false><<<GRID(BN * 4, 512), 128, 0, stream>>>(
      BN * 4, 512, 512, x2h, nullptr, nullptr, 256, 0, idx3, 2, nullptr, nullptr,
      c3Wh, c3b, y3, sD, sD + 512);
  BNFIN(sD, c3g, c3e, 512, 1.f / (float)(BN * 4));
  maxk_kernel<<<(BN * 512 + 255) / 256, 256, 0, stream>>>(
      y3, sD + 2 * 512, sD + 3 * 512, x3h, BN * 512, 2, 9);

  // 4) l1: concat[x1|x2|x3] (896) -> 1024, then global max pool over N
  gemm_bn_wmma<MODE_CONCAT, false><<<GRID(BN, 1024), 128, 0, stream>>>(
      BN, 896, 1024, x1h, x2h, x3h, 128, 256, nullptr, 0, nullptr, nullptr,
      l1Wh, l1b, hb, sE, sE + 1024);
  BNFIN(sE, l1g, l1e, 1024, 1.f / (float)BN);
  maxpool_n_kernel<<<(B * 1024 + 255) / 256, 256, 0, stream>>>(
      hb, sE + 2 * 1024, sE + 3 * 1024, gbh, B * 1024, 10);

  // 5) head: f1 1024->512, f2 512->256, f3 256->1 + sigmoid
  gemm_bn_wmma<MODE_DENSE, false><<<GRID(B, 512), 128, 0, stream>>>(
      B, 1024, 512, gbh, nullptr, nullptr, 0, 0, nullptr, 0, nullptr, nullptr,
      f1Wh, f1b, yf1, sF, sF + 512);
  BNFIN(sF, f1g, f1e, 512, 1.f / (float)B);
  gemm_bn_wmma<MODE_DENSE, true><<<GRID(B, 256), 128, 0, stream>>>(
      B, 512, 256, yf1, nullptr, nullptr, 0, 0, nullptr, 0,
      sF + 2 * 512, sF + 3 * 512, f2Wh, f2b, yf2, sG, sG + 256);
  BNFIN(sG, f2g, f2e, 256, 1.f / (float)B);
  final_fc_kernel<<<1, 64, 0, stream>>>(yf2, sG + 2 * 256, sG + 3 * 256,
                                        f3W, f3b, (float*)d_out, B, 256);
}
